// SwinTransformer_79680233276351
// MI455X (gfx1250) — compile-verified, matrix operand order checked by reading, not run on hardware
//
#include <hip/hip_runtime.h>
#include <hip/hip_bf16.h>
#include <math.h>

typedef __attribute__((ext_vector_type(16))) _Float16 v16h;
typedef __attribute__((ext_vector_type(8)))  float    v8f;

#define WMMA_F16(a, b, c) \
  __builtin_amdgcn_wmma_f32_16x16x32_f16(false, (a), false, (b), (short)0, (c), false, false)

__device__ __forceinline__ int imin(int a, int b) { return a < b ? a : b; }

// Async DMA: copy 16 bytes from global to LDS (per active lane), ASYNCcnt-tracked.
__device__ __forceinline__ void async_b128(unsigned lds_off, const void* gaddr) {
  unsigned long long ga = (unsigned long long)gaddr;
  asm volatile("global_load_async_to_lds_b128 %0, %1, off"
               :: "v"(lds_off), "v"(ga) : "memory");
}
__device__ __forceinline__ void wait_asynccnt0() {
  asm volatile("s_wait_asynccnt 0x0" ::: "memory");
}

// ---------------------------------------------------------------------------
// f32 -> f16 weight convert with K padding: dst[N][Kpad], zero-fill k>=K.
// ---------------------------------------------------------------------------
__global__ void cvt_w_kernel(const float* __restrict__ src, _Float16* __restrict__ dst,
                             int N, int K, int Kpad)
{
  const long i = (long)blockIdx.x * 256 + threadIdx.x;
  if (i >= (long)N * Kpad) return;
  const int n = (int)(i / Kpad), k = (int)(i % Kpad);
  dst[i] = (k < K) ? (_Float16)src[(long)n * K + k] : (_Float16)0.f;
}

// ---------------------------------------------------------------------------
// im2col pack (stride == kernel patches) into f16 [M, Kpad].
// ---------------------------------------------------------------------------
__global__ void im2col_kernel(const float* __restrict__ X, _Float16* __restrict__ A,
                              int Bn, int Cin, int Hin, int ksz, int Kpad, int nchw)
{
  const int Win = Hin, OH = Hin / ksz;
  const long M = (long)Bn * OH * OH;
  const long i = (long)blockIdx.x * 256 + threadIdx.x;
  if (i >= M * Kpad) return;
  const long row = i / Kpad;
  const int gk = (int)(i % Kpad);
  const int K = Cin * ksz * ksz;
  float v = 0.f;
  if (gk < K) {
    const int b = (int)(row / (OH * OH));
    const int pix = (int)(row % (OH * OH));
    const int ho = pix / OH, wo = pix % OH;
    const int ci = gk / (ksz * ksz);
    const int rem = gk % (ksz * ksz);
    const int kh = rem / ksz, kw = rem % ksz;
    const int hi = ho * ksz + kh, wi = wo * ksz + kw;
    v = nchw ? X[(((long)b * Cin + ci) * Hin + hi) * Win + wi]
             : X[((long)b * Hin * Win + (long)hi * Win + wi) * Cin + ci];
  }
  A[i] = (_Float16)v;
}

// ---------------------------------------------------------------------------
// LDS-free WMMA GEMM: C[M,N] = A[M,K]f16 * W[N,K]^T f16 (+bias)(+GELU).
// K multiple of 32. Fragments are direct 32B global loads (contiguous K per
// lane, same k order for A and B). Block 128x128, 8 waves, 32x64/wave.
// Output f16 (Ch) or f32 (Cf).
// ---------------------------------------------------------------------------
__global__ __launch_bounds__(256)
void gemm_tn_wmma(const _Float16* __restrict__ A, const _Float16* __restrict__ W,
                  const float* __restrict__ bias, float* __restrict__ Cf,
                  _Float16* __restrict__ Ch, int M, int N, int K, int act)
{
  const int lane = threadIdx.x & 31, wave = threadIdx.x >> 5;
  const int wm = wave & 3, wn = wave >> 2;
  const int bm = blockIdx.y * 128, bn = blockIdx.x * 128;
  const int r = lane & 15, hf = lane >> 4;

  const _Float16* pA0 = A + (long)imin(bm + wm * 32 + r,      M - 1) * K + hf * 16;
  const _Float16* pA1 = A + (long)imin(bm + wm * 32 + 16 + r, M - 1) * K + hf * 16;
  const _Float16* pB0 = W + (long)imin(bn + wn * 64 + r,      N - 1) * K + hf * 16;
  const _Float16* pB1 = W + (long)imin(bn + wn * 64 + 16 + r, N - 1) * K + hf * 16;
  const _Float16* pB2 = W + (long)imin(bn + wn * 64 + 32 + r, N - 1) * K + hf * 16;
  const _Float16* pB3 = W + (long)imin(bn + wn * 64 + 48 + r, N - 1) * K + hf * 16;

  v8f acc[2][4];
  for (int i = 0; i < 2; i++)
    for (int j = 0; j < 4; j++) acc[i][j] = {};

  for (int k0 = 0; k0 < K; k0 += 32) {
    const v16h a0 = *(const v16h*)(pA0 + k0);
    const v16h a1 = *(const v16h*)(pA1 + k0);
    const v16h b0 = *(const v16h*)(pB0 + k0);
    const v16h b1 = *(const v16h*)(pB1 + k0);
    const v16h b2 = *(const v16h*)(pB2 + k0);
    const v16h b3 = *(const v16h*)(pB3 + k0);
    acc[0][0] = WMMA_F16(a0, b0, acc[0][0]);
    acc[1][0] = WMMA_F16(a1, b0, acc[1][0]);
    acc[0][1] = WMMA_F16(a0, b1, acc[0][1]);
    acc[1][1] = WMMA_F16(a1, b1, acc[1][1]);
    acc[0][2] = WMMA_F16(a0, b2, acc[0][2]);
    acc[1][2] = WMMA_F16(a1, b2, acc[1][2]);
    acc[0][3] = WMMA_F16(a0, b3, acc[0][3]);
    acc[1][3] = WMMA_F16(a1, b3, acc[1][3]);
  }

  for (int i = 0; i < 2; i++)
    for (int j = 0; j < 4; j++) {
#pragma unroll
      for (int e = 0; e < 8; e++) {
        const int row = bm + wm * 32 + i * 16 + e + 8 * hf;
        const int col = bn + wn * 64 + j * 16 + r;
        if (row < M && col < N) {
          float v = acc[i][j][e];
          if (bias) v += bias[col];
          if (act == 1) v = 0.5f * v * (1.f + erff(v * 0.70710678118654752f));
          if (Ch) Ch[(long)row * N + col] = (_Float16)v;
          else    Cf[(long)row * N + col] = v;
        }
      }
    }
}

// ---------------------------------------------------------------------------
// LayerNorm over C per pixel; optional shift-roll + window-partition ordering.
// One wave per row; f32 in (residual stream), f16 out (GEMM operand).
// ---------------------------------------------------------------------------
__global__ __launch_bounds__(256)
void ln_win_kernel(const float* __restrict__ X, const float* __restrict__ g,
                   const float* __restrict__ b, _Float16* __restrict__ O,
                   int Bn, int C, int H, int W, int winmode, int shift)
{
  const long row = (long)blockIdx.x * 8 + (threadIdx.x >> 5);
  const int lane = threadIdx.x & 31;
  const int HW = H * W;
  const long T = (long)Bn * HW;
  if (row >= T) return;

  const int bb = (int)(row / HW);
  const int t  = (int)(row % HW);
  int srcpix;
  if (winmode) {
    const int Wn = W / 7;
    const int w = t / 49, p = t % 49;
    int hh = (w / Wn) * 7 + p / 7;
    int ww = (w % Wn) * 7 + p % 7;
    if (shift) { hh = (hh - 3 + H) % H; ww = (ww - 3 + W) % W; }
    srcpix = hh * W + ww;
  } else {
    srcpix = t;
  }

  const float* xr = X + ((long)bb * HW + srcpix) * C;
  float s = 0.f, s2 = 0.f;
  for (int c = lane; c < C; c += 32) { const float v = xr[c]; s += v; s2 += v * v; }
#pragma unroll
  for (int o = 16; o >= 1; o >>= 1) {
    s  += __shfl_xor(s,  o, 32);
    s2 += __shfl_xor(s2, o, 32);
  }
  const float mu = s / C;
  const float var = s2 / C - mu * mu;
  const float rs = rsqrtf(var + 1e-5f);
  _Float16* orow = O + row * C;
  for (int c = lane; c < C; c += 32)
    orow[c] = (_Float16)((xr[c] - mu) * rs * g[c] + b[c]);
}

// ---------------------------------------------------------------------------
// Fused window attention per (head, window, batch). MM=49 -> 64, Hd=32.
// Q,K tiles DMA'd into LDS via global_load_async_to_lds_b128 (ASYNCcnt),
// scores via WMMA, in-LDS softmax with rel-pos bias + analytic shift mask,
// P*V via WMMA. qkv input f16, output f16.
// ---------------------------------------------------------------------------
__global__ __launch_bounds__(128)
void win_attn_kernel(const _Float16* __restrict__ QKV, const float* __restrict__ bias_table,
                     _Float16* __restrict__ O, int nW, int nH, int H, int W,
                     int shift, float inv_scale)
{
  const int h  = blockIdx.x;
  const int w  = blockIdx.y;
  const int bb = blockIdx.z;
  const int C  = nH * 32;
  const int C3 = 3 * C;
  const long base = ((long)bb * nW + w) * 49;
  const int qoff = h * 96;     // (nH, 3, 32) head offset, in halves

  __shared__ __align__(64) _Float16 sQ[64][32];
  __shared__ __align__(64) _Float16 sK[64][32];
  __shared__ __align__(64) _Float16 sVt[32][64];   // V transposed: [d][key]
  __shared__ __align__(64) float    sS[4][16][64];
  __shared__ __align__(64) _Float16 sP[4][16][64];

  const int tid = threadIdx.x;
  const int lane = tid & 31, wave = tid >> 5;

  // Async DMA of Q and K rows (rows >= 49 left as don't-care: masked later).
  {
    const unsigned ldsQ = (unsigned)(unsigned long long)(const void*)&sQ[0][0];
    const unsigned ldsK = (unsigned)(unsigned long long)(const void*)&sK[0][0];
    const _Float16* qb = QKV + base * (long)C3 + qoff;
    for (int c = tid; c < 256; c += 128) {        // 256 x 16B chunks per tile
      const int row = c >> 2, seg = c & 3;
      if (row < 49) {
        async_b128(ldsQ + (unsigned)(row * 64 + seg * 16), qb + (long)row * C3 + seg * 8);
        async_b128(ldsK + (unsigned)(row * 64 + seg * 16), qb + (long)row * C3 + 32 + seg * 8);
      }
    }
    // V transposed via plain stores (padding keys zeroed so P*V stays finite)
    for (int e = tid; e < 2048; e += 128) {
      const int d = e >> 6, key = e & 63;
      _Float16 v = (_Float16)0.f;
      if (key < 49) v = qb[(long)key * C3 + 64 + d];
      sVt[d][key] = v;
    }
    wait_asynccnt0();
  }
  __syncthreads();

  const int r = lane & 15, hf = lane >> 4;
  const int qr0 = wave * 16;

  // S = q * k^T (one 16x64 strip per wave)
  {
    const v16h aq = *(const v16h*)&sQ[qr0 + r][hf * 16];
    v8f accS[4];
#pragma unroll
    for (int n = 0; n < 4; n++) accS[n] = {};
#pragma unroll
    for (int n = 0; n < 4; n++) {
      const v16h bk = *(const v16h*)&sK[n * 16 + r][hf * 16];
      accS[n] = WMMA_F16(aq, bk, accS[n]);
    }
#pragma unroll
    for (int n = 0; n < 4; n++)
#pragma unroll
      for (int e = 0; e < 8; e++)
        sS[wave][e + 8 * hf][n * 16 + r] = accS[n][e] * inv_scale;
  }
  __syncthreads();

  // softmax with rel-pos bias + shift mask (lane<16: one query row each)
  if (lane < 16) {
    const int q = qr0 + lane;
    const int Wn = W / 7;
    const int wr = w / Wn, wc = w % Wn;
    const int pr = (q < 49) ? q / 7 : 0;
    const int pc = (q < 49) ? q % 7 : 0;
    const int ph = wr * 7 + pr, pw = wc * 7 + pc;
    const int lp = ((ph < H - 7) ? 0 : (ph < H - 3) ? 1 : 2) * 3 +
                   ((pw < W - 7) ? 0 : (pw < W - 3) ? 1 : 2);

    float mx = -1e30f;
    for (int j = 0; j < 64; j++) {
      float v = -1e30f;
      if (j < 49 && q < 49) {
        const int kr = j / 7, kc = j % 7;
        const int idx = (pr - kr + 6) * 13 + (pc - kc + 6);
        v = sS[wave][lane][j] + bias_table[idx * nH + h];
        if (shift) {
          const int kh2 = wr * 7 + kr, kw2 = wc * 7 + kc;
          const int lq = ((kh2 < H - 7) ? 0 : (kh2 < H - 3) ? 1 : 2) * 3 +
                         ((kw2 < W - 7) ? 0 : (kw2 < W - 3) ? 1 : 2);
          if (lq != lp) v = -1e30f;
        }
      }
      sS[wave][lane][j] = v;
      mx = fmaxf(mx, v);
    }
    float sum = 0.f;
    for (int j = 0; j < 64; j++) {
      const float e = __expf(sS[wave][lane][j] - mx);
      sS[wave][lane][j] = e;
      sum += e;
    }
    const float inv = 1.f / sum;
    for (int j = 0; j < 64; j++)
      sP[wave][lane][j] = (_Float16)(sS[wave][lane][j] * inv);
  }
  __syncthreads();

  // O = P * V (M=16, K=64 keys as two 32-steps, N=32 head dims)
  {
    v8f accO[2];
    accO[0] = {}; accO[1] = {};
#pragma unroll
    for (int ks = 0; ks < 2; ks++) {
      const v16h ap = *(const v16h*)&sP[wave][r][ks * 32 + hf * 16];
#pragma unroll
      for (int n = 0; n < 2; n++) {
        const v16h bv = *(const v16h*)&sVt[n * 16 + r][ks * 32 + hf * 16];
        accO[n] = WMMA_F16(ap, bv, accO[n]);
      }
    }
#pragma unroll
    for (int n = 0; n < 2; n++)
#pragma unroll
      for (int e = 0; e < 8; e++) {
        const int rowl = qr0 + e + 8 * hf;
        if (rowl < 49)
          O[(base + rowl) * (long)C + h * 32 + n * 16 + r] = accO[n][e];
      }
  }
}

// ---------------------------------------------------------------------------
// Un-window (faithful linear mapping) + inverse shift + residual add (f32).
// ---------------------------------------------------------------------------
__global__ void unwin_add_kernel(float* __restrict__ X, const float* __restrict__ Y,
                                 int Bn, int C, int H, int W, int shift)
{
  const long i = (long)blockIdx.x * blockDim.x + threadIdx.x;
  const long total = (long)Bn * H * W * C;
  if (i >= total) return;
  const int c = (int)(i % C);
  const long t = i / C;
  const int pix = (int)(t % (H * W));
  const long bb = t / (H * W);
  int sp = pix;
  if (shift) {
    const int hh = pix / W, ww = pix % W;
    sp = ((hh + 3) % H) * W + (ww + 3) % W;
  }
  X[i] += Y[(bb * (long)(H * W) + sp) * C + c];
}

// ---------------------------------------------------------------------------
// Average pool over tokens: O[b,c] = mean_p X[b,p,c]  (f16 -> f16)
// ---------------------------------------------------------------------------
__global__ void pool_kernel(const _Float16* __restrict__ X, _Float16* __restrict__ O,
                            int Bn, int C, int P)
{
  const int i = blockIdx.x * blockDim.x + threadIdx.x;
  if (i >= Bn * C) return;
  const int bb = i / C, c = i % C;
  float s = 0.f;
  for (int p = 0; p < P; p++) s += (float)X[((long)bb * P + p) * C + c];
  O[i] = (_Float16)(s / (float)P);
}

// ---------------------------------------------------------------------------
// Launcher
// ---------------------------------------------------------------------------
static inline void launch_gemm(const _Float16* A, const _Float16* W, const float* bias,
                               float* Cf, _Float16* Ch, int M, int N, int K, int act,
                               hipStream_t s)
{
  dim3 grid((N + 127) / 128, (M + 127) / 128);
  gemm_tn_wmma<<<grid, 256, 0, s>>>(A, W, bias, Cf, Ch, M, N, K, act);
}

extern "C" void kernel_launch(void* const* d_in, const int* in_sizes, int n_in,
                              void* d_out, int out_size, void* d_ws, size_t ws_size,
                              hipStream_t stream)
{
  (void)n_in; (void)out_size; (void)ws_size;
  auto F = [&](int i) -> const float* { return (const float*)d_in[i]; };

  const int Bn = in_sizes[0] / (3 * 224 * 224);   // 16
  const float* x_in   = F(0);
  // jax pytree (sorted dict) order: head_b, head_w, ln_b, ln_g, s0..s3
  const float* head_b = F(1);
  const float* head_w = F(2);
  const float* ln_b_f = F(3);
  const float* ln_g_f = F(4);

  const int depths[4] = {2, 2, 6, 2};
  const int heads[4]  = {3, 6, 12, 24};
  const int cins[4]   = {3, 96, 192, 384};
  const int couts[4]  = {96, 192, 384, 768};
  const int kszs[4]   = {4, 2, 2, 2};
  const int hins[4]   = {224, 56, 28, 14};

  // ---- workspace bump allocator (256B aligned) ----
  size_t off = 0;
  auto alloc = [&](size_t bytes) -> void* {
    off = (off + 255) & ~(size_t)255;
    void* p = (char*)d_ws + off;
    off += bytes;
    return p;
  };
  const long TOKMAX = (long)Bn * 56 * 56 * 96;        // 4,816,896
  float*     xA    = (float*)    alloc(TOKMAX * 4);
  float*     xB    = (float*)    alloc(TOKMAX * 4);
  float*     projT = (float*)    alloc(TOKMAX * 4);
  _Float16*  lnTh  = (_Float16*) alloc(TOKMAX * 2);
  _Float16*  qkvTh = (_Float16*) alloc((long)Bn * 3136 * 288 * 2);
  _Float16*  oTh   = (_Float16*) alloc(TOKMAX * 2);   // attn out / MLP hidden
  _Float16*  im2h  = (_Float16*) alloc(TOKMAX * 2);   // im2col (max 4.82M halves)
  _Float16*  poolh = (_Float16*) alloc((long)Bn * 768 * 2);

  // ---- convert all weights to f16 (K padded to mult of 32) up front ----
  auto cvtW = [&](const float* src, int N, int K, int Kpad) -> const _Float16* {
    _Float16* dst = (_Float16*)alloc((size_t)N * Kpad * 2);
    const long tot = (long)N * Kpad;
    cvt_w_kernel<<<(int)((tot + 255) / 256), 256, 0, stream>>>(src, dst, N, K, Kpad);
    return dst;
  };

  const _Float16* convWh[4];
  const _Float16* qkvWh[12];
  const _Float16* outWh[12];
  const _Float16* fc1Wh[12];
  const _Float16* fc2Wh[12];
  int convKpad[4];
  {
    int base = 5, blkIdx = 0;
    for (int s = 0; s < 4; s++) {
      const int depth = depths[s], C = couts[s];
      const int K = cins[s] * kszs[s] * kszs[s];
      convKpad[s] = (K + 31) & ~31;
      convWh[s] = cvtW(F(base + 13 * depth + 1), C, K, convKpad[s]);
      for (int i = 0; i < depth; i++, blkIdx++) {
        const int bk = base + 13 * i;
        fc1Wh[blkIdx] = cvtW(F(bk + 2),  C,     C, C);
        fc2Wh[blkIdx] = cvtW(F(bk + 4),  C,     C, C);
        outWh[blkIdx] = cvtW(F(bk + 10), C,     C, C);
        qkvWh[blkIdx] = cvtW(F(bk + 12), 3 * C, C, C);
      }
      base += 13 * depth + 2;
    }
  }
  const _Float16* headWh = cvtW(head_w, 1000, 768, 768);

  // ---- forward ----
  float* cur = xA;
  float* nxt = xB;
  int base = 5, blkIdx = 0;
  for (int s = 0; s < 4; s++) {
    const int depth = depths[s];
    const float* conv_b = F(base + 13 * depth);
    const int Cin = cins[s], Cout = couts[s], k = kszs[s], Hin = hins[s];
    const int H = Hin / k, W = H;
    const int M = Bn * H * W;
    {
      const int Kpad = convKpad[s];
      const long tot = (long)M * Kpad;
      im2col_kernel<<<(int)((tot + 255) / 256), 256, 0, stream>>>(
          (s == 0) ? x_in : cur, im2h, Bn, Cin, Hin, k, Kpad, (s == 0) ? 1 : 0);
      launch_gemm(im2h, convWh[s], conv_b, nxt, nullptr, M, Cout, Kpad, 0, stream);
    }
    { float* t = cur; cur = nxt; nxt = t; }

    const int C = Cout;
    const int nW = (H / 7) * (W / 7);
    const int nH = heads[s];
    const int T = Bn * H * W;
    const float inv_scale = 1.0f / sqrtf((float)C * (float)C / (float)nH);

    for (int i = 0; i < depth; i++, blkIdx++) {
      const int bk = base + 13 * i;
      const float* bias_table = F(bk + 0);
      const float* fc1_b = F(bk + 1);
      const float* fc2_b = F(bk + 3);
      const float* ln1_b = F(bk + 5);
      const float* ln1_g = F(bk + 6);
      const float* ln2_b = F(bk + 7);
      const float* ln2_g = F(bk + 8);
      const float* out_b = F(bk + 9);
      const float* qkv_b = F(bk + 11);
      const int shift = (i & 1);

      // LN1 + shift-roll + window-partition (window-ordered f16 tokens)
      ln_win_kernel<<<(T + 7) / 8, 256, 0, stream>>>(cur, ln1_g, ln1_b, lnTh,
                                                     Bn, C, H, W, 1, shift);
      // qkv projection (f16 out)
      launch_gemm(lnTh, qkvWh[blkIdx], qkv_b, nullptr, qkvTh, T, 3 * C, C, 0, stream);
      // fused window attention (async Q/K DMA + WMMA + softmax + WMMA)
      win_attn_kernel<<<dim3(nH, nW, Bn), 128, 0, stream>>>(
          qkvTh, bias_table, oTh, nW, nH, H, W, shift, inv_scale);
      // output projection (f32 out)
      launch_gemm(oTh, outWh[blkIdx], out_b, projT, nullptr, T, C, C, 0, stream);
      // un-window + inverse roll + residual
      {
        const long tot = (long)T * C;
        unwin_add_kernel<<<(int)((tot + 255) / 256), 256, 0, stream>>>(
            cur, projT, Bn, C, H, W, shift);
      }
      // MLP: LN2 -> fc1(+GELU, f16) -> fc2(f32) -> residual
      ln_win_kernel<<<(T + 7) / 8, 256, 0, stream>>>(cur, ln2_g, ln2_b, lnTh,
                                                     Bn, C, H, W, 0, 0);
      launch_gemm(lnTh, fc1Wh[blkIdx], fc1_b, nullptr, oTh, T, C, C, 1, stream);
      launch_gemm(oTh,  fc2Wh[blkIdx], fc2_b, projT, nullptr, T, C, C, 0, stream);
      {
        const long tot = (long)T * C;
        unwin_add_kernel<<<(int)((tot + 255) / 256), 256, 0, stream>>>(
            cur, projT, Bn, C, H, W, 0);
      }
    }
    base += 13 * depth + 2;
  }

  // final LN -> global average pool -> classifier head
  {
    const int C = 768, H = 7, W = 7;
    const int T = Bn * H * W;
    ln_win_kernel<<<(T + 7) / 8, 256, 0, stream>>>(cur, ln_g_f, ln_b_f, lnTh,
                                                   Bn, C, H, W, 0, 0);
    pool_kernel<<<(Bn * C + 255) / 256, 256, 0, stream>>>(lnTh, poolh, Bn, C, 49);
    launch_gemm(poolh, headWh, head_b, (float*)d_out, nullptr, Bn, 1000, C, 0, stream);
  }
}
